// Attention_12799002542580
// MI455X (gfx1250) — compile-verified, matrix operand order checked by reading, not run on hardware
//
#include <hip/hip_runtime.h>

// ---------------------------------------------------------------------------
// Types for CDNA5 WMMA (wave32): 16x16x32 bf16 -> f32
// ---------------------------------------------------------------------------
typedef __bf16  v16bf __attribute__((ext_vector_type(16)));
typedef float   v8f   __attribute__((ext_vector_type(8)));
typedef unsigned int v4u __attribute__((ext_vector_type(4)));
typedef int     v8i   __attribute__((ext_vector_type(8)));
typedef int     v4i   __attribute__((ext_vector_type(4)));

#define WMMA_BF16(a, b, c) \
    __builtin_amdgcn_wmma_f32_16x16x32_bf16(false, (a), false, (b), (short)0, (c), false, false)

#if __has_builtin(__builtin_amdgcn_tensor_load_to_lds) && \
    __has_builtin(__builtin_amdgcn_s_wait_tensorcnt)
#define HAVE_TDM 1
#else
#define HAVE_TDM 0
#endif

static constexpr int B_  = 8;
static constexpr int S_  = 1024;
static constexpr int H_  = 16;
static constexpr int HS_ = 64;
static constexpr int D_  = 1024;
static constexpr int M_  = B_ * S_;   // 8192 rows

#if HAVE_TDM
// ---------------------------------------------------------------------------
// TDM: 2-D tile DMA  global(bf16, row stride = kdim elems) -> LDS (compacted).
// D# layout per CDNA5 ISA ch.8: group0 = {count/type, lds_addr, global_addr},
// group1 = {data_size, tensor dims, tile dims, dim0 stride}. data_size=1 (2B).
// ---------------------------------------------------------------------------
__device__ __forceinline__ void tdm_load_tile_bf16(unsigned int lds_addr,
                                                   const void* gptr,
                                                   int kdim,      // elements per row
                                                   int trows,     // tensor rows
                                                   int tile_rows) // tile rows (tile k = 32)
{
    unsigned long long ga = (unsigned long long)gptr;
    v4u g0 = { 1u,                                     // count=1, user mode
               lds_addr,                               // LDS byte address
               (unsigned int)ga,                       // global_addr[31:0]
               (unsigned int)((ga >> 32) & 0x1FFFFFFu) | 0x80000000u }; // [56:32] | type=2
    v8i g1 = { (int)0x00010000u,                                   // data_size=1 (2B)
               (int)(((unsigned)kdim & 0xFFFFu) << 16),            // tensor_dim0 lo
               (int)((((unsigned)kdim >> 16) & 0xFFFFu) |
                     (((unsigned)trows & 0xFFFFu) << 16)),         // dim0 hi | dim1 lo
               (int)((((unsigned)trows >> 16) & 0xFFFFu) |
                     (32u << 16)),                                 // dim1 hi | tile_dim0=32
               (int)((unsigned)tile_rows & 0xFFFFu),               // tile_dim1 (tile_dim2=0)
               (int)(unsigned)kdim,                                // tensor_dim0_stride lo32
               0, 0 };
    v4i z4 = {0, 0, 0, 0};
#if defined(__clang_major__) && (__clang_major__ >= 23)
    v8i z8 = {0, 0, 0, 0, 0, 0, 0, 0};
    __builtin_amdgcn_tensor_load_to_lds(g0, g1, z4, z4, z8, 0);
#else
    __builtin_amdgcn_tensor_load_to_lds(g0, g1, z4, z4, 0);
#endif
}

__device__ __forceinline__ unsigned int lds_addr_of(const void* p) {
    // generic LDS pointer: addr[31:0] is the LDS byte offset (flat aperture rule)
    return (unsigned int)(unsigned long long)p;
}
#endif // HAVE_TDM

// ---------------------------------------------------------------------------
// Elementwise fp32 -> bf16 convert (x)
// ---------------------------------------------------------------------------
__global__ __launch_bounds__(256) void cvt_f32_bf16(const float* __restrict__ in,
                                                    __bf16* __restrict__ out, int n) {
    int i = (blockIdx.x * 256 + threadIdx.x) * 4;
    if (i + 3 < n) {
        float4 v = *(const float4*)(in + i);
        out[i + 0] = (__bf16)v.x;
        out[i + 1] = (__bf16)v.y;
        out[i + 2] = (__bf16)v.z;
        out[i + 3] = (__bf16)v.w;
    }
}

// ---------------------------------------------------------------------------
// Transpose-convert weight: WT[n][k] = (bf16) W[k][n]   (1024x1024)
// ---------------------------------------------------------------------------
__global__ __launch_bounds__(256) void transpose_cvt(const float* __restrict__ W,
                                                     __bf16* __restrict__ WT) {
    __shared__ float tile[32][33];
    int x  = blockIdx.x * 32 + threadIdx.x;   // column of W
    int y0 = blockIdx.y * 32;                 // row block of W
    for (int i = threadIdx.y; i < 32; i += 8)
        tile[i][threadIdx.x] = W[(size_t)(y0 + i) * D_ + x];
    __syncthreads();
    int xt = y0 + threadIdx.x;                // column of WT == row of W
    for (int i = threadIdx.y; i < 32; i += 8)
        WT[(size_t)(blockIdx.x * 32 + i) * D_ + xt] = (__bf16)tile[threadIdx.x][i];
}

// ---------------------------------------------------------------------------
// bf16 GEMM: C[M][1024] = A[M][K] @ BT[1024][K]^T + bias, WMMA 16x16x32 bf16.
// Block: 256 threads = 8 waves. Block tile 128(M) x 128(N). K step 32.
// Operand staging: TDM double-buffered async DMA (A 128x32, B 128x32 tiles)
// issued by wave 0, pipelined with s_wait_tensorcnt; fallback = cooperative
// global->LDS copy when the TDM builtin is unavailable.
// mode 0: store bf16 row-major            (Q, K projections)
// mode 1: store bf16 head-transposed Vt[B][H][HS][S]   (V projection)
// mode 2: store fp32 row-major, += residual (out projection, y = x + ctx@Wo+bo)
// ---------------------------------------------------------------------------
__global__ __launch_bounds__(256) void gemm_bf16(const __bf16* __restrict__ A,
                                                 const __bf16* __restrict__ BT,
                                                 const float* __restrict__ bias,
                                                 const float* __restrict__ residual,
                                                 void* __restrict__ Cout,
                                                 int Kdim, int mode) {
    __shared__ __attribute__((aligned(32))) __bf16 As[2][128 * 32];   // 2 x 8 KB
    __shared__ __attribute__((aligned(32))) __bf16 Bs[2][128 * 32];   // 2 x 8 KB

    const int tid  = threadIdx.x;
    const int wave = tid >> 5;
    const int lane = tid & 31;
    const int col  = lane & 15;       // B/C column within 16x16 tile
    const int half = lane >> 4;       // K-half (A/B) or M-half (C)
    const int n0   = blockIdx.x * 128;
    const int m0   = blockIdx.y * 128;

    const v8f vzero = {0.f, 0.f, 0.f, 0.f, 0.f, 0.f, 0.f, 0.f};
    v8f acc[8];
#pragma unroll
    for (int i = 0; i < 8; ++i) acc[i] = vzero;

#if HAVE_TDM
    const bool issuer = (tid < 32);   // wave 0 drives the TDM engine
    if (issuer) {
        tdm_load_tile_bf16(lds_addr_of(&As[0][0]), A  + (size_t)m0 * Kdim, Kdim, M_,   128);
        tdm_load_tile_bf16(lds_addr_of(&Bs[0][0]), BT + (size_t)n0 * Kdim, Kdim, 1024, 128);
    }
    int buf = 0;
    for (int k0 = 0; k0 < Kdim; k0 += 32) {
        if (issuer) {
            if (k0 + 32 < Kdim) {
                tdm_load_tile_bf16(lds_addr_of(&As[buf ^ 1][0]),
                                   A + (size_t)m0 * Kdim + (k0 + 32), Kdim, M_, 128);
                tdm_load_tile_bf16(lds_addr_of(&Bs[buf ^ 1][0]),
                                   BT + (size_t)n0 * Kdim + (k0 + 32), Kdim, 1024, 128);
                __builtin_amdgcn_s_wait_tensorcnt(2);  // in-order: first pair landed
            } else {
                __builtin_amdgcn_s_wait_tensorcnt(0);
            }
        }
        __syncthreads();   // publish tile buf to all waves

        v16bf af = *(const v16bf*)(&As[buf][(wave * 16 + col) * 32 + half * 16]);
#pragma unroll
        for (int nt = 0; nt < 8; ++nt) {
            v16bf bf = *(const v16bf*)(&Bs[buf][(nt * 16 + col) * 32 + half * 16]);
            acc[nt] = WMMA_BF16(af, bf, acc[nt]);
        }
        __syncthreads();   // all reads done before wave0 overwrites buf next iter
        buf ^= 1;
    }
#else
    const int arow = tid >> 1;        // 0..127 staging row
    const int ah   = tid & 1;         // 16-element half of the 32-wide k-slab
    for (int k0 = 0; k0 < Kdim; k0 += 32) {
        {
            const uint4* srcA = (const uint4*)(A  + (size_t)(m0 + arow) * Kdim + k0 + ah * 16);
            const uint4* srcB = (const uint4*)(BT + (size_t)(n0 + arow) * Kdim + k0 + ah * 16);
            uint4* dstA = (uint4*)(&As[0][arow * 32 + ah * 16]);
            uint4* dstB = (uint4*)(&Bs[0][arow * 32 + ah * 16]);
            dstA[0] = srcA[0]; dstA[1] = srcA[1];
            dstB[0] = srcB[0]; dstB[1] = srcB[1];
            if (k0 + 32 < Kdim)
                __builtin_prefetch(A + (size_t)(m0 + arow) * Kdim + k0 + 32, 0, 1);
        }
        __syncthreads();
        v16bf af = *(const v16bf*)(&As[0][(wave * 16 + col) * 32 + half * 16]);
#pragma unroll
        for (int nt = 0; nt < 8; ++nt) {
            v16bf bf = *(const v16bf*)(&Bs[0][(nt * 16 + col) * 32 + half * 16]);
            acc[nt] = WMMA_BF16(af, bf, acc[nt]);
        }
        __syncthreads();
    }
#endif

    // ---- epilogue ----
#pragma unroll
    for (int nt = 0; nt < 8; ++nt) {
        const int n = n0 + nt * 16 + col;
        const float bs = bias[n];
#pragma unroll
        for (int r = 0; r < 8; ++r) {
            const int m = m0 + wave * 16 + r + half * 8;
            float v = acc[nt][r] + bs;
            if (mode == 0) {
                ((__bf16*)Cout)[(size_t)m * D_ + n] = (__bf16)v;
            } else if (mode == 1) {
                const int b = m >> 10, s = m & 1023;
                const int h = n >> 6,  hs = n & 63;
                ((__bf16*)Cout)[((((size_t)b * H_ + h) * HS_ + hs) * S_) + s] = (__bf16)v;
            } else {
                v += residual[(size_t)m * D_ + n];
                ((float*)Cout)[(size_t)m * D_ + n] = v;
            }
        }
    }
}

// ---------------------------------------------------------------------------
// Attention: one wave (32 threads) per (b, h, 16-query strip).
// LDS: 16x1024 fp32 scores (64KB) + 16x1024 bf16 probs (32KB) + row stats.
// Pass1: scores = Q K^T / 8 + mask-bias  (2 WMMAs per key tile, 64 tiles)
// Pass1.5: per-row max & sum-exp (lane-parallel + shfl_xor across halves)
// Pass2: write probs to global attn (single HBM touch), bf16 probs to LDS,
//        then ctx = P @ V with 4 WMMAs per 32-key chunk (Vt head-transposed).
// ---------------------------------------------------------------------------
__global__ __launch_bounds__(32) void attention_kernel(const __bf16* __restrict__ Qb,
                                                       const __bf16* __restrict__ Kb,
                                                       const __bf16* __restrict__ Vt,
                                                       const int*    __restrict__ mask,
                                                       float*  __restrict__ attn_out,
                                                       __bf16* __restrict__ ctxb) {
    extern __shared__ __attribute__((aligned(32))) char smem[];
    float*  sc   = (float*)smem;                       // 16*1024 f32  = 64KB
    __bf16* pb   = (__bf16*)(smem + 65536);            // 16*1024 bf16 = 32KB
    float*  mrow = (float*)(smem + 65536 + 32768);     // 16 f32
    float*  irow = mrow + 16;                          // 16 f32

    const int lane = threadIdx.x;
    const int col  = lane & 15;
    const int half = lane >> 4;
    const int q0   = blockIdx.x * 16;
    const int h    = blockIdx.y;
    const int b    = blockIdx.z;

    const size_t qbase = ((size_t)b * S_ + q0 + col) * D_ + h * HS_;
    v16bf a0 = *(const v16bf*)(Qb + qbase + half * 16);        // hs 0..31
    v16bf a1 = *(const v16bf*)(Qb + qbase + 32 + half * 16);   // hs 32..63

    const float scale = 0.125f;   // 1/sqrt(64)

    // ---- Pass 1: scores into LDS ----
    for (int kt = 0; kt < 64; ++kt) {
        const int key = kt * 16 + col;
        const size_t kb = ((size_t)b * S_ + key) * D_ + h * HS_;
        v16bf b0 = *(const v16bf*)(Kb + kb + half * 16);
        v16bf b1 = *(const v16bf*)(Kb + kb + 32 + half * 16);
        v8f c = {0.f, 0.f, 0.f, 0.f, 0.f, 0.f, 0.f, 0.f};
        c = WMMA_BF16(a0, b0, c);
        c = WMMA_BF16(a1, b1, c);
        const float biasl = -10000.0f * (float)mask[b * S_ + key];
#pragma unroll
        for (int r = 0; r < 8; ++r) {
            float s = c[r] * scale + biasl;
            sc[(r + half * 8) * 1024 + kt * 16 + col] = s;
        }
    }
    __syncthreads();

    // ---- Pass 1.5: row max and sum of exp ----
    {
        const float* p = sc + col * 1024 + half * 512;
        float m = -3.0e38f;
        for (int i = 0; i < 512; ++i) m = fmaxf(m, p[i]);
        m = fmaxf(m, __shfl_xor(m, 16, 32));
        float s = 0.f;
        for (int i = 0; i < 512; ++i) s += __expf(p[i] - m);
        s += __shfl_xor(s, 16, 32);
        if (half == 0) { mrow[col] = m; irow[col] = 1.0f / s; }
    }
    __syncthreads();

    // ---- Pass 2a: probs -> global attn + LDS bf16 ----
    const size_t abase = (((size_t)b * H_ + h) * S_ + q0) * S_;
    for (int i = lane; i < 16 * 1024; i += 32) {
        const int row = i >> 10;
        const float p = __expf(sc[i] - mrow[row]) * irow[row];
        attn_out[abase + i] = p;
        pb[i] = (__bf16)p;
    }
    __syncthreads();

    // ---- Pass 2b: ctx = P @ V  (Vt layout [B][H][HS][S]) ----
    const v8f vzero = {0.f, 0.f, 0.f, 0.f, 0.f, 0.f, 0.f, 0.f};
    v8f acc[4];
    acc[0] = vzero; acc[1] = vzero; acc[2] = vzero; acc[3] = vzero;
    const size_t vbase = ((size_t)b * H_ + h) * HS_ * S_;
    for (int k2 = 0; k2 < 32; ++k2) {
        v16bf af = *(const v16bf*)(pb + col * 1024 + k2 * 32 + half * 16);
#pragma unroll
        for (int nt = 0; nt < 4; ++nt) {
            const int hs = nt * 16 + col;
            v16bf bf = *(const v16bf*)(Vt + vbase + (size_t)hs * S_ + k2 * 32 + half * 16);
            acc[nt] = WMMA_BF16(af, bf, acc[nt]);
        }
    }
#pragma unroll
    for (int nt = 0; nt < 4; ++nt) {
#pragma unroll
        for (int r = 0; r < 8; ++r) {
            const int q = q0 + r + half * 8;
            ctxb[((size_t)b * S_ + q) * D_ + h * HS_ + nt * 16 + col] = (__bf16)acc[nt][r];
        }
    }
}

// ---------------------------------------------------------------------------
// Row LayerNorm: one block per row of y[M][1024]
// ---------------------------------------------------------------------------
__global__ __launch_bounds__(256) void layernorm_kernel(const float* __restrict__ y,
                                                        const float* __restrict__ gamma,
                                                        const float* __restrict__ beta,
                                                        float* __restrict__ out) {
    const int row = blockIdx.x;
    const float* p = y + (size_t)row * D_;
    __shared__ float rs[8], rss[8];

    float s = 0.f, ss = 0.f;
    for (int i = threadIdx.x; i < D_; i += 256) {
        const float v = p[i];
        s += v; ss += v * v;
    }
#pragma unroll
    for (int off = 16; off; off >>= 1) {
        s  += __shfl_xor(s, off, 32);
        ss += __shfl_xor(ss, off, 32);
    }
    const int wave = threadIdx.x >> 5, lane = threadIdx.x & 31;
    if (lane == 0) { rs[wave] = s; rss[wave] = ss; }
    __syncthreads();
    if (wave == 0) {
        s  = (lane < 8) ? rs[lane]  : 0.f;
        ss = (lane < 8) ? rss[lane] : 0.f;
#pragma unroll
        for (int off = 4; off; off >>= 1) {
            s  += __shfl_xor(s, off, 32);
            ss += __shfl_xor(ss, off, 32);
        }
        if (lane == 0) { rs[0] = s; rss[0] = ss; }
    }
    __syncthreads();
    const float mean = rs[0] * (1.0f / (float)D_);
    const float var  = rss[0] * (1.0f / (float)D_) - mean * mean;
    const float inv  = rsqrtf(var + 1e-6f);
    for (int i = threadIdx.x; i < D_; i += 256)
        out[(size_t)row * D_ + i] = (p[i] - mean) * inv * gamma[i] + beta[i];
}

// ---------------------------------------------------------------------------
// Host-side orchestration
// ---------------------------------------------------------------------------
extern "C" void kernel_launch(void* const* d_in, const int* in_sizes, int n_in,
                              void* d_out, int out_size, void* d_ws, size_t ws_size,
                              hipStream_t stream) {
    (void)in_sizes; (void)n_in; (void)out_size; (void)ws_size;

    const float* x     = (const float*)d_in[0];
    const int*   mask  = (const int*)  d_in[1];
    const float* wq    = (const float*)d_in[2];
    const float* bq    = (const float*)d_in[3];
    const float* wk    = (const float*)d_in[4];
    const float* bk    = (const float*)d_in[5];
    const float* wv    = (const float*)d_in[6];
    const float* bv    = (const float*)d_in[7];
    const float* wo    = (const float*)d_in[8];
    const float* bo    = (const float*)d_in[9];
    const float* gamma = (const float*)d_in[10];
    const float* beta  = (const float*)d_in[11];

    float* out_y  = (float*)d_out;                         // [B,S,D]  8.4M f32
    float* attn_o = out_y + (size_t)M_ * D_;               // [B,H,S,S] 134M f32

    // Workspace carve (bytes)
    char* ws = (char*)d_ws;
    const size_t MB = 1024 * 1024;
    __bf16* xb   = (__bf16*)(ws);              // 16 MB
    __bf16* wqT  = (__bf16*)(ws + 16 * MB);    //  2 MB each
    __bf16* wkT  = (__bf16*)(ws + 18 * MB);
    __bf16* wvT  = (__bf16*)(ws + 20 * MB);
    __bf16* woT  = (__bf16*)(ws + 22 * MB);
    __bf16* Qb   = (__bf16*)(ws + 24 * MB);    // 16 MB
    __bf16* Kb   = (__bf16*)(ws + 40 * MB);    // 16 MB
    __bf16* Vt   = (__bf16*)(ws + 56 * MB);    // 16 MB  [B][H][HS][S]
    __bf16* ctxb = (__bf16*)(ws + 72 * MB);    // 16 MB
    float*  yres = (float*) (ws + 88 * MB);    // 32 MB  x + ctx@Wo + bo

    // 1) convert activations
    cvt_f32_bf16<<<(M_ * D_) / (256 * 4), 256, 0, stream>>>(x, xb, M_ * D_);

    // 2) transpose-convert weights
    dim3 tgrid(32, 32), tblk(32, 8);
    transpose_cvt<<<tgrid, tblk, 0, stream>>>(wq, wqT);
    transpose_cvt<<<tgrid, tblk, 0, stream>>>(wk, wkT);
    transpose_cvt<<<tgrid, tblk, 0, stream>>>(wv, wvT);
    transpose_cvt<<<tgrid, tblk, 0, stream>>>(wo, woT);

    // 3) QKV projections (block tile 128x128)
    dim3 ggrid(D_ / 128, M_ / 128);
    gemm_bf16<<<ggrid, 256, 0, stream>>>(xb, wqT, bq, nullptr, (void*)Qb, D_, 0);
    gemm_bf16<<<ggrid, 256, 0, stream>>>(xb, wkT, bk, nullptr, (void*)Kb, D_, 0);
    gemm_bf16<<<ggrid, 256, 0, stream>>>(xb, wvT, bv, nullptr, (void*)Vt, D_, 1);

    // 4) attention (scores+softmax+attn-write+ctx) : 96KB+stats dynamic LDS
    dim3 agrid(S_ / 16, H_, B_);
    const size_t asmem = 64 * 1024 + 32 * 1024 + 32 * sizeof(float);
    attention_kernel<<<agrid, 32, asmem, stream>>>(Qb, Kb, Vt, mask, attn_o, ctxb);

    // 5) output projection + residual
    gemm_bf16<<<ggrid, 256, 0, stream>>>(ctxb, woT, bo, x, (void*)yres, D_, 2);

    // 6) layernorm
    layernorm_kernel<<<M_, 256, 0, stream>>>(yres, gamma, beta, out_y);
}
